// DenoiseNet_8607114461272
// MI455X (gfx1250) — matrix operand order repaired and checked by reading, not compiled, not run on hardware
//
#include <hip/hip_runtime.h>

typedef _Float16 v16h __attribute__((ext_vector_type(16)));
typedef _Float16 v8h  __attribute__((ext_vector_type(8)));
typedef float    v8f  __attribute__((ext_vector_type(8)));

#define HH 512
#define WW 512

// ---------------- LDS layout (bytes) for main kernel ----------------
#define XS_OFF    0
#define XS_N     (4*36*36)                  // f32  x tile + halo2, 4th channel zeros (pads im2col K to 32)
#define FEAT_OFF (XS_OFF + XS_N*4)          // 20736
#define FEAT_N   (1168*64)                  // f16  feat [pixel][channel], 34x34 region padded to 1168 px
#define SEK_OFF  (FEAT_OFF + FEAT_N*2)      // 170240
#define SEK_N    (512*64)                   // f16  sek [pixel][channel] for one 16-row half-tile
#define WB_OFF   (SEK_OFF + SEK_N*2)        // 235776
#define WB_N     (32*64)                    // f16  w_feat im2col [k32][c64]
#define WK_OFF   (WB_OFF + WB_N*2)          // 239872
#define WK_N     (64*32)                    // f16  w_k [c64][o32]
#define WSA_OFF  (WK_OFF + WK_N*2)          // 243968  f16 [tap9][c64] (padded)
#define CASA_OFF (WSA_OFF + 1280)           // 245248  f16[64]  ca[b][c]+b_sa[c]
#define BK_OFF   (CASA_OFF + 128)           // 245376  f32[32]  b_k (padded)
#define BF_OFF   (BK_OFF + 128)             // 245504  f32[64]  b_feat
#define KERN_OFF (BF_OFF + 256)             // 245760
#define KERN_N   (8*32*16)                  // f32  per-wave kernels [wave][o32][16px]
#define SMEM_BYTES (KERN_OFF + KERN_N*4)    // 262144 bytes = 256KB < 320KB WGP LDS

// im2col offset into the 4-channel padded xs plane for K index k (k<36)
__host__ __device__ constexpr int koff(int k) {
    return (k / 9) * 1296 + ((k % 9) / 3) * 36 + ((k % 9) % 3);
}

// ---------------- Pass 1a: shifted plane sums of x ----------------
__global__ void k_shift_sums(const float* __restrict__ x, float* __restrict__ shifted) {
    int b  = blockIdx.x / 3;
    int ci = blockIdx.x % 3;
    const float* p = x + ((size_t)(b * 3 + ci)) * HH * WW;

    float s = 0.f, r0 = 0.f, r1 = 0.f, c0 = 0.f, c1 = 0.f;
    for (int i = threadIdx.x; i < HH * WW; i += blockDim.x) {
        float v = p[i];
        int iy = i >> 9, ix = i & (WW - 1);
        s += v;
        if (iy == 0)      r0 += v;
        if (iy == HH - 1) r1 += v;
        if (ix == 0)      c0 += v;
        if (ix == WW - 1) c1 += v;
    }
    __shared__ float red[256];
    auto blockReduce = [&](float v) -> float {
        red[threadIdx.x] = v; __syncthreads();
        for (int st = 128; st > 0; st >>= 1) {
            if (threadIdx.x < (unsigned)st) red[threadIdx.x] += red[threadIdx.x + st];
            __syncthreads();
        }
        float r = red[0]; __syncthreads(); return r;
    };
    s  = blockReduce(s);
    r0 = blockReduce(r0);
    r1 = blockReduce(r1);
    c0 = blockReduce(c0);
    c1 = blockReduce(c1);

    if (threadIdx.x == 0) {
        float x00 = p[0], x0w = p[WW - 1];
        float xh0 = p[(size_t)(HH - 1) * WW], xhw = p[(size_t)(HH - 1) * WW + WW - 1];
        for (int dy = 0; dy < 3; ++dy) {
            for (int dx = 0; dx < 3; ++dx) {
                float v = s;
                float rexc = (dy == 0) ? r1 : (dy == 2 ? r0 : 0.f);
                float cexc = (dx == 0) ? c1 : (dx == 2 ? c0 : 0.f);
                v -= rexc + cexc;
                if (dy == 0 && dx == 0) v += xhw;
                if (dy == 0 && dx == 2) v += xh0;
                if (dy == 2 && dx == 0) v += x0w;
                if (dy == 2 && dx == 2) v += x00;
                shifted[(b * 3 + ci) * 9 + dy * 3 + dx] = v;
            }
        }
    }
}

// ---------------- Pass 1b: analytic mean(feat) -> ECA -> ca[b][c] ----------------
__global__ void k_eca(const float* __restrict__ shifted,
                      const float* __restrict__ w_feat, const float* __restrict__ b_feat,
                      const float* __restrict__ w_ca, const float* __restrict__ b_ca,
                      float* __restrict__ ca_out) {
    int b = blockIdx.x;
    int c = threadIdx.x;           // 64 threads
    __shared__ float m[66];
    float acc = b_feat[c] * (float)(HH * WW);
    for (int ci = 0; ci < 3; ++ci)
        for (int t = 0; t < 9; ++t)
            acc += w_feat[c * 27 + ci * 9 + t] * shifted[(b * 3 + ci) * 9 + t];
    m[c + 1] = acc * (1.f / (float)(HH * WW));
    if (c == 0) { m[0] = 0.f; m[65] = 0.f; }
    __syncthreads();
    ca_out[b * 64 + c] = b_ca[0] + w_ca[0] * m[c] + w_ca[1] * m[c + 1] + w_ca[2] * m[c + 2];
}

// ---------------- Main fused kernel: 32x32 tile per workgroup ----------------
__global__ void __launch_bounds__(256)
k_main(const float* __restrict__ x,
       const float* __restrict__ w_feat, const float* __restrict__ b_feat,
       const float* __restrict__ w_sa,  const float* __restrict__ b_sa,
       const float* __restrict__ w_k,   const float* __restrict__ b_k,
       const float* __restrict__ ca_in, float* __restrict__ out) {
    extern __shared__ char smem[];
    float*    xs    = (float*)(smem + XS_OFF);      // [4][36][36]
    _Float16* featL = (_Float16*)(smem + FEAT_OFF); // [1168][64]
    _Float16* sekL  = (_Float16*)(smem + SEK_OFF);  // [512][64]
    _Float16* wB    = (_Float16*)(smem + WB_OFF);   // [32][64]
    _Float16* wK    = (_Float16*)(smem + WK_OFF);   // [64][32]
    _Float16* wsa2  = (_Float16*)(smem + WSA_OFF);  // [9][64]
    _Float16* casaH = (_Float16*)(smem + CASA_OFF); // [64]
    float*    bkL   = (float*)(smem + BK_OFF);      // [32]
    float*    bfL   = (float*)(smem + BF_OFF);      // [64]
    float*    kernL = (float*)(smem + KERN_OFF);    // [8][32][16]

    const int tid  = threadIdx.x;
    const int lane = tid & 31;
    const int wv   = tid >> 5;       // wave id 0..7
    const int half = lane >> 4;
    const int lr   = lane & 15;

    const int b  = blockIdx.z;
    const int oy = blockIdx.y * 32;
    const int ox = blockIdx.x * 32;

    // ---- stage weights / biases into LDS ----
    for (int i = tid; i < 32 * 64; i += 256) {         // wB[k][c]
        int k = i >> 6, c = i & 63;
        wB[i] = (_Float16)((k < 27) ? w_feat[c * 27 + k] : 0.f);
    }
    for (int i = tid; i < 64 * 32; i += 256) {         // wK[c][o]
        int c = i >> 5, o = i & 31;
        wK[i] = (_Float16)((o < 27) ? w_k[o * 64 + c] : 0.f);
    }
    for (int i = tid; i < 9 * 64; i += 256) {          // wsa2[t][c]
        int t = i >> 6, c = i & 63;
        wsa2[i] = (_Float16)w_sa[c * 9 + t];
    }
    if (tid < 64) {
        casaH[tid] = (_Float16)(ca_in[b * 64 + tid] + b_sa[tid]);
        bfL[tid]   = b_feat[tid];
    }
    if (tid < 32) bkL[tid] = (tid < 27) ? b_k[tid] : 0.f;

    // ---- zero x region (incl. dummy 4th channel), then async-load valid pairs ----
    for (int i = tid; i < XS_N; i += 256) xs[i] = 0.f;
    __syncthreads();
    for (int i2 = tid; i2 < 3 * 36 * 18; i2 += 256) {  // b64 pairs (region x is even-aligned)
        int ci  = i2 / (36 * 18);
        int rem = i2 - ci * 36 * 18;
        int ry  = rem / 18, rx = (rem - ry * 18) * 2;
        int iy  = oy - 2 + ry, ix = ox - 2 + rx;
        if (iy >= 0 && iy < HH && ix >= 0 && ix + 1 < WW) {
            size_t gidx = ((size_t)(b * 3 + ci) * HH + iy) * WW + ix;
            int lidx = ci * 1296 + ry * 36 + rx;
#if defined(__gfx1250__)
            unsigned ldsAddr = (unsigned)(size_t)(&xs[lidx]);
            const float* gp = &x[gidx];
            asm volatile("global_load_async_to_lds_b64 %0, %1, off"
                         :: "v"(ldsAddr), "v"(gp) : "memory");
#else
            xs[lidx] = x[gidx]; xs[lidx + 1] = x[gidx + 1];
#endif
        }
    }
#if defined(__gfx1250__)
    asm volatile("s_wait_asynccnt 0" ::: "memory");
#endif
    __syncthreads();

    // ---- B fragments: w_feat (4 N-tiles) and w_k (2 K-blocks x 2 N-tiles) ----
    v16h bf[4];
#pragma unroll
    for (int nt = 0; nt < 4; ++nt)
#pragma unroll
        for (int e = 0; e < 16; ++e)
            bf[nt][e] = wB[(8 * half + e + (e >= 8 ? 8 : 0)) * 64 + nt * 16 + lr];
    v16h bk2[2][2];
#pragma unroll
    for (int kb = 0; kb < 2; ++kb)
#pragma unroll
        for (int nt = 0; nt < 2; ++nt)
#pragma unroll
            for (int e = 0; e < 16; ++e)
                bk2[kb][nt][e] = wK[(kb * 32 + 8 * half + e + (e >= 8 ? 8 : 0)) * 32 + nt * 16 + lr];

    float bias4[4] = { bfL[lr], bfL[16 + lr], bfL[32 + lr], bfL[48 + lr] };
    float bko0 = bkL[lr], bko1 = bkL[16 + lr];

    // ================= Phase 1: feat = conv3x3(x) via WMMA =================
    for (int mt = wv; mt < 73; mt += 8) {
        int mbase = mt * 16;
        int p = mbase + lr; if (p > 1155) p = 1155;
        int fy = p / 34, fx = p - fy * 34;
        int base = fy * 36 + fx;
        // two-pass: 16 independent ds_load_b32 first, packed cvt after
        float tmp[16];
#pragma unroll
        for (int e = 0; e < 16; ++e) {
            int klo  = e + (e >= 8 ? 8 : 0);
            int off0 = koff(klo);
            int off1 = koff(klo + 8);
            tmp[e] = xs[base + (half ? off1 : off0)];
        }
        v16h a;
#pragma unroll
        for (int e = 0; e < 16; ++e) a[e] = (_Float16)tmp[e];
#pragma unroll
        for (int nt = 0; nt < 4; ++nt) {
            float bv = bias4[nt];
            v8f acc = {bv,bv,bv,bv,bv,bv,bv,bv};      // bias via C init
            acc = __builtin_amdgcn_wmma_f32_16x16x32_f16(
                      false, a, false, bf[nt], (short)0, acc, false, false);
            int c = nt * 16 + lr;
#pragma unroll
            for (int r = 0; r < 8; ++r)
                featL[(mbase + 8 * half + r) * 64 + c] = (_Float16)acc[r];
        }
    }
    __syncthreads();

    // ============ Phase 2 per 16-row half: depthwise sek -> WMMA -> dyn filter ============
    for (int h = 0; h < 2; ++h) {
        // ---- Phase 2a: sek[p][c] via packed f16 depthwise conv (2 px per thread) ----
        {
            int r   = tid >> 4;            // 0..15 rows of half
            int tx  = (tid & 15) * 2;      // even col
            int ty  = h * 16 + r;
            int phx = r * 32 + tx;         // pixel index within half
#pragma unroll
            for (int ch = 0; ch < 8; ++ch) {
                const int cb = ch * 8;
                v8h w9[9];
#pragma unroll
                for (int t = 0; t < 9; ++t)
                    w9[t] = *(const v8h*)(wsa2 + t * 64 + cb);
                v8h cinit = *(const v8h*)(casaH + cb);
                v8h acc0 = cinit, acc1 = cinit;
#pragma unroll
                for (int dy = 0; dy < 3; ++dy) {
                    v8h row[4];
#pragma unroll
                    for (int dx = 0; dx < 4; ++dx)
                        row[dx] = *(const v8h*)(featL + ((ty + dy) * 34 + tx + dx) * 64 + cb);
#pragma unroll
                    for (int dx = 0; dx < 3; ++dx) {
                        acc0 += row[dx]     * w9[dy * 3 + dx];
                        acc1 += row[dx + 1] * w9[dy * 3 + dx];
                    }
                }
                *(v8h*)(sekL + phx * 64 + cb)       = acc0;
                *(v8h*)(sekL + (phx + 1) * 64 + cb) = acc1;
            }
        }
        __syncthreads();

        // ---- Phase 2b: per-pixel kernels via WMMA + dynamic filtering ----
        for (int q = 0; q < 4; ++q) {
            int mt = wv * 4 + q;               // 32 m-tiles of 16 px over the half
            int mbase = mt * 16;
            int ph = mbase + lr;               // A row pixel (half-local)

            // A fragments: two contiguous 8-channel runs per lane -> 2x ds_load_b128 each
            v8h lo0 = *(const v8h*)(sekL + ph * 64 + 8 * half);
            v8h hi0 = *(const v8h*)(sekL + ph * 64 + 16 + 8 * half);
            v8h lo1 = *(const v8h*)(sekL + ph * 64 + 32 + 8 * half);
            v8h hi1 = *(const v8h*)(sekL + ph * 64 + 48 + 8 * half);
            v16h a0 = __builtin_shufflevector(lo0, hi0, 0,1,2,3,4,5,6,7,8,9,10,11,12,13,14,15);
            v16h a1 = __builtin_shufflevector(lo1, hi1, 0,1,2,3,4,5,6,7,8,9,10,11,12,13,14,15);

            v8f k0 = {bko0,bko0,bko0,bko0,bko0,bko0,bko0,bko0};   // b_k via C init
            v8f k1 = {bko1,bko1,bko1,bko1,bko1,bko1,bko1,bko1};
            k0 = __builtin_amdgcn_wmma_f32_16x16x32_f16(false, a0, false, bk2[0][0], (short)0, k0, false, false);
            k0 = __builtin_amdgcn_wmma_f32_16x16x32_f16(false, a1, false, bk2[1][0], (short)0, k0, false, false);
            k1 = __builtin_amdgcn_wmma_f32_16x16x32_f16(false, a0, false, bk2[0][1], (short)0, k1, false, false);
            k1 = __builtin_amdgcn_wmma_f32_16x16x32_f16(false, a1, false, bk2[1][1], (short)0, k1, false, false);

            // stage kernels [o][px]: each lane's 8 pixels contiguous -> 4x ds_store_b128
            {
                float4 s0 = {k0[0], k0[1], k0[2], k0[3]};
                float4 s1 = {k0[4], k0[5], k0[6], k0[7]};
                float4 s2 = {k1[0], k1[1], k1[2], k1[3]};
                float4 s3 = {k1[4], k1[5], k1[6], k1[7]};
                float* kp0 = kernL + (wv * 32 + lr) * 16 + 8 * half;
                float* kp1 = kernL + (wv * 32 + 16 + lr) * 16 + 8 * half;
                *(float4*)(kp0)     = s0;
                *(float4*)(kp0 + 4) = s1;
                *(float4*)(kp1)     = s2;
                *(float4*)(kp1 + 4) = s3;
            }
#if defined(__gfx1250__)
            asm volatile("s_wait_dscnt 0" ::: "memory");   // same-wave LDS RAW fence
#else
            __syncthreads();
#endif
            // dynamic filtering: 48 tasks (16 px x 3 out channels) over 32 lanes
            for (int t = lane; t < 48; t += 32) {
                int pl = t & 15, cc = t >> 4;
                int pp = mbase + pl;                  // half-local pixel
                int ty2 = h * 16 + (pp >> 5), tx2 = pp & 31;
                const float* kp = kernL + (wv * 32 + cc * 9) * 16 + pl;
                const float* xp = xs + cc * 1296 + (ty2 + 1) * 36 + (tx2 + 1);
                float o = 0.f;
#pragma unroll
                for (int dy = 0; dy < 3; ++dy)
#pragma unroll
                    for (int dx = 0; dx < 3; ++dx)
                        o += kp[(dy * 3 + dx) * 16] * xp[dy * 36 + dx];
                out[((size_t)(b * 3 + cc) * HH + (oy + ty2)) * WW + (ox + tx2)] = o;
            }
        }
        __syncthreads();   // protect sekL before next half overwrites it
    }
}

extern "C" void kernel_launch(void* const* d_in, const int* in_sizes, int n_in,
                              void* d_out, int out_size, void* d_ws, size_t ws_size,
                              hipStream_t stream) {
    (void)in_sizes; (void)n_in; (void)out_size; (void)ws_size;
    const float* x      = (const float*)d_in[0];
    const float* w_feat = (const float*)d_in[1];
    const float* b_feat = (const float*)d_in[2];
    const float* w_sa   = (const float*)d_in[3];
    const float* b_sa   = (const float*)d_in[4];
    const float* w_ca   = (const float*)d_in[5];
    const float* b_ca   = (const float*)d_in[6];
    const float* w_k    = (const float*)d_in[7];
    const float* b_k    = (const float*)d_in[8];
    float* out = (float*)d_out;

    float* ws_shifted = (float*)d_ws;                    // [8][3][9]
    float* ws_ca      = (float*)((char*)d_ws + 1024);    // [8][64]

    k_shift_sums<<<24, 256, 0, stream>>>(x, ws_shifted);
    k_eca<<<8, 64, 0, stream>>>(ws_shifted, w_feat, b_feat, w_ca, b_ca, ws_ca);
    k_main<<<dim3(16, 16, 8), 256, SMEM_BYTES, stream>>>(
        x, w_feat, b_feat, w_sa, b_sa, w_k, b_k, ws_ca, out);
}
